// HHNet_62397284876290
// MI455X (gfx1250) — matrix operand order repaired and checked by reading, not compiled, not run on hardware
//
#include <hip/hip_runtime.h>

// ---------------------------------------------------------------------------
// Fused HH spiking-net forward for MI455X (gfx1250, wave32, WMMA).
//
// Two-crew software pipeline per workgroup (one workgroup per batch element):
//   waves 4..7 ("GEMM crew"): z1 chunk i via v_wmma_f32_16x16x32_f16
//       waves 4..6: two adjacent N-tiles each (branch-free, reuse_a hint)
//       wave  7   : last N-tile
//       (B fragments pre-swizzled in LDS -> 2x ds_load_b128 per fragment)
//   waves 0..3 ("scan crew"): sequential HH scan of chunk i-1 (100 lanes),
//       wave 0: z2 WMMA + layer-2 scan + output store + T1 carry,
//       waves 1..3: stage/convert chunk i+1 activations + prefetch.
// Double-buffered sA / sZ1; 2 barriers per 16-step chunk.
// ---------------------------------------------------------------------------

typedef __attribute__((ext_vector_type(16))) _Float16 v16h;
typedef __attribute__((ext_vector_type(8)))  float    v8f;

#define DTC   0.01f
#define EK    (-77.0f)
#define EL    (-65.0f)
#define ENA   55.0f
#define IAPP  1.5f
#define KP    3.0f
#define VT    20.0f
#define VS    0.0f
#define AD    1.0f
#define AR    0.1f
#define GK    35.0f
#define GL    0.3f
#define GNA   40.0f
#define GSY   0.04f

#define BATCH 100
#define TLEN  2000
#define NIN   784
#define N1    100
#define N2    10

#define KPAD1 800          // 784 -> multiple of 32
#define NPAD1 112          // 100 -> multiple of 16
#define KPAD2 128          // 100 -> multiple of 32
#define CHUNK 16
#define NCHUNKS (TLEN / CHUNK)   // 125
#define NT1   7            // N-tiles of 16 covering 112
#define NKC1  (KPAD1 / 32) // 25 K-chunks, GEMM1
#define NKC2  (KPAD2 / 32) // 4  K-chunks, GEMM2
#define KHALF 416          // 13 of 25 K-chunks in pipeline segment A

__device__ __forceinline__ float safe_div(float num, float den, float fb) {
    return (den == 0.0f) ? fb : num / den;     // jnp.where(den==0, fb, num/den)
}
__device__ __forceinline__ float gate(float a, float b, float prev) {
    const float s = 0.5f * DTC * (a + b);
    return (a * DTC + (1.0f - s) * prev) / (s + 1.0f);
}
__device__ __forceinline__ float sigmoidf(float x) {
    return 1.0f / (1.0f + __expf(-x));
}
__device__ __forceinline__ void hh_step(float& V, float& m, float& n, float& h,
                                        float& y, float zk) {
    const float p1 = GNA * m * m * m * h;
    const float p2 = GK * n * n * n * n;
    const float Gs = 0.5f * DTC * (p1 + p2 + GL + GSY * y);
    const float E  = p1 * ENA + p2 * EK + GL * EL + GSY * VS * y;
    const float Vn = (V * (1.0f - Gs) + DTC * (E + IAPP)) / (1.0f + Gs);
    const float aN = safe_div( 0.02f  * (Vn - 25.0f), 1.0f - __expf((25.0f - Vn) / 9.0f), 0.18f );
    const float bN = safe_div(-0.002f * (Vn - 25.0f), 1.0f - __expf((Vn - 25.0f) / 9.0f), 0.018f);
    const float aM = safe_div( 0.182f * (Vn + 35.0f), 1.0f - __expf((-Vn - 35.0f) / 9.0f), 1.638f);
    const float bM = safe_div(-0.124f * (Vn + 35.0f), 1.0f - __expf((Vn + 35.0f) / 9.0f), 1.116f);
    const float aH = 0.25f * __expf((-Vn - 90.0f) / 12.0f);
    const float bH = 0.25f * __expf((Vn + 34.0f) / 12.0f);
    m = gate(aM, bM, m);
    n = gate(aN, bN, n);
    h = gate(aH, bH, h);
    y = gate(AD * zk, AR, y);
    V = Vn;
}

__global__ __launch_bounds__(256, 1)
void hhnet_fused(const float* __restrict__ x,     // (B, T, 784)
                 const float* __restrict__ W1,    // (100, 784)
                 const float* __restrict__ W2,    // (10, 100)
                 float* __restrict__ out)         // (B, T, 10)
{
    const int b    = blockIdx.x;
    const int tid  = threadIdx.x;
    const int lane = tid & 31;
    const int wv   = tid >> 5;
    const int crew = (wv >= 4);          // 0 = scan crew, 1 = GEMM crew
    const int wvg  = wv - 4;             // GEMM-crew wave index 0..3

    // ---- LDS (~249 KB of the 320 KB CDNA5 per-WGP budget) ----
    __shared__ v16h     sW1f[NT1 * NKC1 * 32];    // 179,200 B (B-fragment order)
    __shared__ v16h     sW2f[NKC2 * 32];          //   4,096 B (B-fragment order)
    __shared__ _Float16 sA [2][CHUNK][KPAD1];     //  51,200 B (double buffer)
    __shared__ float    sZ1[2][CHUNK][NPAD1];     //  14,336 B (double buffer)
    __shared__ _Float16 sT1[CHUNK + 1][KPAD2];    //   4,352 B (row0 = carry)
    __shared__ float    sZ2[CHUNK][16];           //   1,024 B
    __shared__ float    sOut[CHUNK][16];          //   1,024 B

    const size_t xbase = (size_t)b * TLEN * NIN;

    // ---- one-time: swizzle W1/W2 into CDNA5 B-fragment order (f16) ----
    // fragment f = (nt*NKC1 + kcs)*32 + lane ; element e:
//   N = nt*16 + (lane&15), K = kcs*32 + (lane>>4)*8 + e + (e>=8 ? 8 : 0)
    for (int f = tid; f < NT1 * NKC1 * 32; f += 256) {
        const int l   = f & 31;
        const int kcs = (f >> 5) % NKC1;
        const int nt  = (f >> 5) / NKC1;
        const int n   = nt * 16 + (l & 15);
        const int kb  = kcs * 32 + (l >> 4) * 8;
        v16h v;
#pragma unroll
        for (int e = 0; e < 16; ++e) {
            const int k = kb + e + (e >= 8 ? 8 : 0);
            v[e] = (_Float16)((k < NIN && n < N1) ? W1[n * NIN + k] : 0.0f);
        }
        sW1f[f] = v;
    }
    for (int f = tid; f < NKC2 * 32; f += 256) {
        const int l   = f & 31;
        const int kcs = f >> 5;
        const int j   = l & 15;
        const int kb  = kcs * 32 + (l >> 4) * 8;
        v16h v;
#pragma unroll
        for (int e = 0; e < 16; ++e) {
            const int k = kb + e + (e >= 8 ? 8 : 0);
            v[e] = (_Float16)((k < N1 && j < N2) ? W2[j * N1 + k] : 0.0f);
        }
        sW2f[f] = v;
    }
    // T1 carry row (time 0) + zero padding columns
    const float T1init = sigmoidf((-70.0f - VT) / KP);
    for (int idx = tid; idx < (CHUNK + 1) * KPAD2; idx += 256) {
        const int r = idx / KPAD2, nn = idx % KPAD2;
        sT1[r][nn] = (_Float16)((r == 0 && nn < N1) ? T1init : 0.0f);
    }
    // t = 0 output row (V2 init = -70)
    if (tid < N2)
        out[((size_t)b * TLEN + 0) * N2 + tid] = T1init;

    // ---- prologue: stage chunk 0 activations (all threads) ----
    for (int idx = tid; idx < CHUNK * KPAD1; idx += 256) {
        const int r = idx / KPAD1, k = idx % KPAD1;
        sA[0][r][k] = (k < NIN) ? (_Float16)x[xbase + (size_t)r * NIN + k]
                                : (_Float16)0.0f;
    }

    // ---- per-lane neuron state ----
    float V1 = -70.0f, m1 = 0.0f, n1 = 0.0f, h1 = 1.0f, y1 = 0.0f;  // tid < 100
    float V2 = -70.0f, m2 = 0.0f, n2 = 0.0f, h2 = 1.0f, y2 = 1.0f;  // lane < 10, wv==0

    // ======================= pipelined chunk loop =======================
    for (int i = 0; i <= NCHUNKS; ++i) {
        const bool haveG = (i < NCHUNKS);
        const bool haveS = (i > 0);
        const int  gbuf  = i & 1;

        __syncthreads();   // ---- barrier A ----

        v8f acc0 = {}, acc1 = {};
        const int half = lane >> 4;
        const int row  = lane & 15;

        // ================= segment A =================
        if (crew && haveG) {
            if (wvg < 3) {
                // waves 4..6: adjacent tiles 2*wvg, 2*wvg+1 (branch-free)
                const int f0 = (2 * wvg)     * NKC1 * 32 + lane;
                const int f1 = (2 * wvg + 1) * NKC1 * 32 + lane;
                for (int kc = 0; kc < KHALF; kc += 32) {
                    v16h a;
                    const int kb = kc + half * 8;
#pragma unroll
                    for (int e = 0; e < 8; ++e) {          // A frag: 2x16B runs
                        a[e]     = sA[gbuf][row][kb + e];
                        a[e + 8] = sA[gbuf][row][kb + 16 + e];
                    }
                    const int ko = (kc >> 5) * 32;
                    acc0 = __builtin_amdgcn_wmma_f32_16x16x32_f16(
                               false, a, false, sW1f[f0 + ko],
                               (short)0, acc0, false, false);
                    acc1 = __builtin_amdgcn_wmma_f32_16x16x32_f16(
                               false, a, false, sW1f[f1 + ko],
                               (short)0, acc1, /*reuse_a=*/true, false);
                }
            } else {
                // wave 7: tile 6
                const int f0 = 6 * NKC1 * 32 + lane;
                for (int kc = 0; kc < KHALF; kc += 32) {
                    v16h a;
                    const int kb = kc + half * 8;
#pragma unroll
                    for (int e = 0; e < 8; ++e) {
                        a[e]     = sA[gbuf][row][kb + e];
                        a[e + 8] = sA[gbuf][row][kb + 16 + e];
                    }
                    acc0 = __builtin_amdgcn_wmma_f32_16x16x32_f16(
                               false, a, false, sW1f[f0 + (kc >> 5) * 32],
                               (short)0, acc0, false, false);
                }
            }
        } else if (!crew && haveS && tid < N1) {
            // layer-1 scan: 16 sequential HH steps for chunk i-1
            const int ts = (i - 1) * CHUNK;
            const float* z1row = &sZ1[(i - 1) & 1][0][0];
#pragma unroll 1
            for (int k = 0; k < CHUNK; ++k) {
                const int t = ts + k + 1;
                if (t < TLEN) {
                    hh_step(V1, m1, n1, h1, y1, z1row[k * NPAD1 + tid]);
                    sT1[k + 1][tid] = (_Float16)sigmoidf((V1 - VT) / KP);
                }
            }
        }

        __syncthreads();   // ---- barrier B ----

        // ================= segment B =================
        if (crew && haveG) {
            const int mb = half * 8;                       // C/D: M = r + half*8
            if (wvg < 3) {
                const int f0 = (2 * wvg)     * NKC1 * 32 + lane;
                const int f1 = (2 * wvg + 1) * NKC1 * 32 + lane;
                for (int kc = KHALF; kc < KPAD1; kc += 32) {
                    v16h a;
                    const int kb = kc + half * 8;
#pragma unroll
                    for (int e = 0; e < 8; ++e) {
                        a[e]     = sA[gbuf][row][kb + e];
                        a[e + 8] = sA[gbuf][row][kb + 16 + e];
                    }
                    const int ko = (kc >> 5) * 32;
                    acc0 = __builtin_amdgcn_wmma_f32_16x16x32_f16(
                               false, a, false, sW1f[f0 + ko],
                               (short)0, acc0, false, false);
                    acc1 = __builtin_amdgcn_wmma_f32_16x16x32_f16(
                               false, a, false, sW1f[f1 + ko],
                               (short)0, acc1, /*reuse_a=*/true, false);
                }
#pragma unroll
                for (int r = 0; r < 8; ++r) {
                    sZ1[gbuf][mb + r][(2 * wvg)     * 16 + row] = acc0[r];
                    sZ1[gbuf][mb + r][(2 * wvg + 1) * 16 + row] = acc1[r];
                }
            } else {
                const int f0 = 6 * NKC1 * 32 + lane;
                for (int kc = KHALF; kc < KPAD1; kc += 32) {
                    v16h a;
                    const int kb = kc + half * 8;
#pragma unroll
                    for (int e = 0; e < 8; ++e) {
                        a[e]     = sA[gbuf][row][kb + e];
                        a[e + 8] = sA[gbuf][row][kb + 16 + e];
                    }
                    acc0 = __builtin_amdgcn_wmma_f32_16x16x32_f16(
                               false, a, false, sW1f[f0 + (kc >> 5) * 32],
                               (short)0, acc0, false, false);
                }
#pragma unroll
                for (int r = 0; r < 8; ++r)
                    sZ1[gbuf][mb + r][6 * 16 + row] = acc0[r];
            }
        } else if (!crew && wv == 0 && haveS) {
            // ---- wave 0: z2 WMMA + layer-2 scan + output (chunk i-1) ----
            const int ts = (i - 1) * CHUNK;
            v8f c2 = {};
#pragma unroll
            for (int kc = 0; kc < KPAD2; kc += 32) {
                v16h a;
                const int kb = kc + half * 8;
#pragma unroll
                for (int e = 0; e < 8; ++e) {
                    a[e]     = sT1[row][kb + e];
                    a[e + 8] = sT1[row][kb + 16 + e];
                }
                c2 = __builtin_amdgcn_wmma_f32_16x16x32_f16(
                         false, a, false, sW2f[(kc >> 5) * 32 + lane],
                         (short)0, c2, false, false);
            }
            const int mb = half * 8;
#pragma unroll
            for (int r = 0; r < 8; ++r)
                sZ2[mb + r][row] = c2[r];

            if (lane < N2) {
#pragma unroll 1
                for (int k = 0; k < CHUNK; ++k) {
                    const int t = ts + k + 1;
                    if (t < TLEN) {
                        hh_step(V2, m2, n2, h2, y2, sZ2[k][lane]);
                        sOut[k][lane] = sigmoidf((V2 - VT) / KP);
                    }
                }
            }
            // coalesced output store (intra-wave LDS deps -> s_wait_dscnt)
            for (int j = lane; j < CHUNK * N2; j += 32) {
                const int k = j / N2, jj = j % N2;
                const int t = ts + k + 1;
                if (t < TLEN)
                    out[((size_t)b * TLEN + t) * N2 + jj] = sOut[k][jj];
            }
            // T1 carry: row 16 -> row 0 (after GEMM2 consumed row 0)
            for (int j = lane; j < KPAD2; j += 32)
                sT1[0][j] = sT1[CHUNK][j];
        } else if (!crew && wv > 0 && (i + 1) < NCHUNKS) {
            // ---- waves 1..3: stage + convert chunk i+1 into sA[(i+1)&1] ----
            const int tnext = (i + 1) * CHUNK;
            _Float16 (*An)[KPAD1] = sA[(i + 1) & 1];
            for (int idx = tid - 32; idx < CHUNK * KPAD1; idx += 96) {
                const int r = idx / KPAD1, k = idx % KPAD1;
                An[r][k] = (k < NIN)
                             ? (_Float16)x[xbase + (size_t)(tnext + r) * NIN + k]
                             : (_Float16)0.0f;
            }
            // prefetch chunk i+2 (256B lines) from wave 1
            if (wv == 1 && (i + 2) < NCHUNKS) {
                const float* nx = &x[xbase + (size_t)(i + 2) * CHUNK * NIN];
                for (int l2 = lane; l2 < (CHUNK * NIN * 4) / 256; l2 += 32)
                    __builtin_prefetch(nx + l2 * 64, 0, 0);
            }
        }
    }
}

extern "C" void kernel_launch(void* const* d_in, const int* in_sizes, int n_in,
                              void* d_out, int out_size, void* d_ws, size_t ws_size,
                              hipStream_t stream) {
    const float* x  = (const float*)d_in[0];   // (100, 2000, 784) f32
    const float* W1 = (const float*)d_in[1];   // (100, 784)       f32
    const float* W2 = (const float*)d_in[2];   // (10, 100)        f32
    float* out = (float*)d_out;                // (100, 2000, 10)  f32
    (void)in_sizes; (void)n_in; (void)out_size; (void)d_ws; (void)ws_size;
    hipLaunchKernelGGL(hhnet_fused, dim3(BATCH), dim3(256), 0, stream,
                       x, W1, W2, out);
}
